// Attention_69870527971950
// MI455X (gfx1250) — matrix operand order repaired and checked by reading, not compiled, not run on hardware
//
#include <hip/hip_runtime.h>
#include <hip/hip_bf16.h>

#define B_  4
#define T_  2048
#define NX_ 1024
#define NH_ 16
#define HD_ 64
#define MTOK (B_*T_)               // 8192
#define BHTD (B_*NH_*T_*HD_)       // 8388608

typedef __attribute__((ext_vector_type(16))) __bf16 v16bf;
typedef __attribute__((ext_vector_type(8)))  float  v8f;
typedef __attribute__((ext_vector_type(4)))  unsigned int u32x4;
typedef __attribute__((ext_vector_type(8)))  int i32x8;
typedef __attribute__((ext_vector_type(4)))  int i32x4;

union Frag { v16bf v; u32x4 q[2]; };

__device__ __forceinline__ unsigned short f2bf(float f) {
    union { float f; unsigned u; } x; x.f = f;
    unsigned u = x.u;
    unsigned r = u + 0x7FFFu + ((u >> 16) & 1u);
    return (unsigned short)(r >> 16);
}

__global__ void cvt_f32_to_bf16(const float* __restrict__ src,
                                unsigned short* __restrict__ dst, int n) {
    int i = blockIdx.x * blockDim.x + threadIdx.x;
    int stride = gridDim.x * blockDim.x;
    for (; i < n; i += stride) dst[i] = f2bf(src[i]);
}

// wt[n][k] = bf16(w[k][n]); K fixed at 1024. Writes coalesced, reads hit L2.
__global__ void transpose_cvt(const float* __restrict__ w,
                              unsigned short* __restrict__ wt, int N, int total) {
    int idx = blockIdx.x * blockDim.x + threadIdx.x;
    int stride = gridDim.x * blockDim.x;
    for (; idx < total; idx += stride) {
        int n = idx >> 10;
        int k = idx & 1023;
        wt[idx] = f2bf(w[(size_t)k * N + n]);
    }
}

// ---------------------------------------------------------------------------
// Tensor Data Mover: 2D tile load, D# built per CDNA5 ISA
// ---------------------------------------------------------------------------
#if defined(__has_builtin)
#if __has_builtin(__builtin_amdgcn_tensor_load_to_lds)
#define HAVE_TDM 1
#endif
#endif

#ifdef HAVE_TDM
__device__ __forceinline__ void tdm_load_2d(unsigned lds_addr, const void* gptr,
                                            unsigned g1dw0,        // data_size/pad cfg
                                            unsigned td0, unsigned td1,
                                            unsigned tile0, unsigned tile1,
                                            unsigned stride0) {
    unsigned long long ga = (unsigned long long)(uintptr_t)gptr;
    u32x4 g0 = { 1u,                                   // count=1, valid user D#
                 lds_addr,                             // lds_addr (bytes)
                 (unsigned)ga,                         // global_addr[95:64]
                 ((unsigned)(ga >> 32) & 0x01FFFFFFu) | (2u << 30) };  // addr hi | type=2
    i32x8 g1 = { (int)g1dw0,
                 (int)(td0 << 16),                     // tensor_dim0[15:0]
                 (int)((td0 >> 16) | (td1 << 16)),     // td0 hi | td1 lo
                 (int)((td1 >> 16) | (tile0 << 16)),   // td1 hi | tile_dim0
                 (int)tile1,                           // tile_dim1 (tile_dim2=0)
                 (int)stride0,                         // tensor_dim0_stride lo
                 0, 0 };
    i32x4 gz = { 0, 0, 0, 0 };
#if defined(__clang_major__) && __clang_major__ >= 23
    i32x8 gz8 = { 0, 0, 0, 0, 0, 0, 0, 0 };
    __builtin_amdgcn_tensor_load_to_lds(g0, g1, gz, gz, gz8, 0);
#else
    __builtin_amdgcn_tensor_load_to_lds(g0, g1, gz, gz, 0);
#endif
}
// group1 dw0 = data_size(2B)=1<<16 | pad_enable<<20 | pad_interval<<22 | pad_amount<<25
#define TDM_DW0_K  (0x10000u | (1u << 20) | (4u << 22) | (3u << 25))  // 32DW rows, +4DW pad -> pitch 72
#define TDM_DW0_V  (0x10000u | (1u << 20) | (3u << 22) | (3u << 25))  // 16DW rows, +4DW pad -> pitch 40
#endif

// ===========================================================================
// GEMM core: 64(M) x 128(N) x 32(K) block tile, 8 waves (2x4), each wave a
// 32x32 macro-tile = 4 WMMAs per K-step from 8 ds_load_b128.
// Both operands staged with pure b128 copies (weights pre-transposed [N,K]).
// ===========================================================================
#define GEMM_PROLOGUE(XA, WT, WROW)                                            \
    __shared__ __align__(16) unsigned short sA[64 * 48];                       \
    __shared__ __align__(16) unsigned short sBt[128 * 48];                     \
    const int tid = threadIdx.x;                                               \
    const int w = tid >> 5, l = tid & 31;                                      \
    const int wy = w >> 2, wx = w & 3;                                         \
    const int hh = l >> 4, ln = l & 15;                                        \
    const int m0 = blockIdx.y * 64;                                            \
    const int n0 = blockIdx.x * 128;                                           \
    v8f acc00 = {}, acc01 = {}, acc10 = {}, acc11 = {};                        \
    const int arow = tid >> 2, acg = tid & 3;                                  \
    const int brow = tid >> 1, bkg = tid & 1;                                  \
    for (int k0 = 0; k0 < 1024; k0 += 32) {                                    \
        __syncthreads();                                                       \
        u32x4 av = *(const u32x4*)(XA + (size_t)(m0 + arow) * 1024 + k0 + acg * 8); \
        *(u32x4*)(sA + arow * 48 + acg * 8) = av;                              \
        const unsigned short* wsrc = WT + (size_t)(n0 + brow) * WROW + k0 + bkg * 16; \
        u32x4 bv0 = *(const u32x4*)(wsrc);                                     \
        u32x4 bv1 = *(const u32x4*)(wsrc + 8);                                 \
        *(u32x4*)(sBt + brow * 48 + bkg * 16) = bv0;                           \
        *(u32x4*)(sBt + brow * 48 + bkg * 16 + 8) = bv1;                       \
        __syncthreads();                                                       \
        Frag fb0, fb1, fa0, fa1;                                               \
        {                                                                      \
            const unsigned short* p = sBt + (wx * 32 + ln) * 48 + 8 * hh;      \
            fb0.q[0] = *(const u32x4*)p;                                       \
            fb0.q[1] = *(const u32x4*)(p + 16);                                \
            const unsigned short* p2 = sBt + (wx * 32 + 16 + ln) * 48 + 8 * hh; \
            fb1.q[0] = *(const u32x4*)p2;                                      \
            fb1.q[1] = *(const u32x4*)(p2 + 16);                               \
        }                                                                      \
        {                                                                      \
            const unsigned short* p = sA + (wy * 32 + ln) * 48;                \
            fa0.q[0] = *(const u32x4*)(p + 8 * hh);                            \
            fa0.q[1] = *(const u32x4*)(p + 16 + 8 * hh);                       \
            const unsigned short* p2 = sA + (wy * 32 + 16 + ln) * 48;          \
            fa1.q[0] = *(const u32x4*)(p2 + 8 * hh);                           \
            fa1.q[1] = *(const u32x4*)(p2 + 16 + 8 * hh);                      \
        }                                                                      \
        acc00 = __builtin_amdgcn_wmma_f32_16x16x32_bf16(false, fa0.v, false, fb0.v, (short)0, acc00, false, false); \
        acc01 = __builtin_amdgcn_wmma_f32_16x16x32_bf16(false, fa0.v, false, fb1.v, (short)0, acc01, false, false); \
        acc10 = __builtin_amdgcn_wmma_f32_16x16x32_bf16(false, fa1.v, false, fb0.v, (short)0, acc10, false, false); \
        acc11 = __builtin_amdgcn_wmma_f32_16x16x32_bf16(false, fa1.v, false, fb1.v, (short)0, acc11, false, false); \
    }

// ---------------------------------------------------------------------------
// QKV GEMM: scatter epilogue -> q_bf [B,H,T,D], present k/v fp32,
// k_bf [B,H,T,D], vt_bf [B,H,D,T] (pre-transposed for attention staging)
// ---------------------------------------------------------------------------
__global__ __launch_bounds__(256) void qkv_gemm(
    const unsigned short* __restrict__ xa,     // [8192,1024] bf16
    const unsigned short* __restrict__ wt,     // [3072,1024] bf16 (transposed)
    const float* __restrict__ bias,            // [3072]
    unsigned short* __restrict__ qb,
    unsigned short* __restrict__ kb,
    unsigned short* __restrict__ vt,
    float* __restrict__ present)               // [2,B,H,T,D]
{
    GEMM_PROLOGUE(xa, wt, 1024)

    #pragma unroll
    for (int sn = 0; sn < 2; ++sn) {
        const int n_g  = n0 + wx * 32 + sn * 16 + ln;
        const float bi = bias[n_g];
        const int sect = n_g >> 10;
        const int col  = n_g & 1023;
        const int head = col >> 6, d = col & 63;
        #pragma unroll
        for (int sm = 0; sm < 2; ++sm) {
            v8f acc = sm ? (sn ? acc11 : acc10) : (sn ? acc01 : acc00);
            #pragma unroll
            for (int r = 0; r < 8; ++r) {
                int m_g = m0 + wy * 32 + sm * 16 + r + 8 * hh;
                float val = acc[r] + bi;
                int bb = m_g >> 11, tt = m_g & 2047;
                size_t hidx = ((size_t)((bb * NH_ + head) * T_ + tt)) * HD_ + d;
                if (sect == 0) {
                    qb[hidx] = f2bf(val);
                } else if (sect == 1) {
                    present[hidx] = val;
                    kb[hidx] = f2bf(val);
                } else {
                    present[(size_t)BHTD + hidx] = val;
                    vt[((size_t)((bb * NH_ + head) * HD_ + d)) * T_ + tt] = f2bf(val);
                }
            }
        }
    }
}

// ---------------------------------------------------------------------------
// Flash attention: grid (B*NH, T/128). 8 waves x 16 query rows each.
// K/V tiles staged by the Tensor Data Mover (wave 0 issues, TENSORcnt +
// workgroup barrier completes); V already transposed [d][t] in memory, TDM
// padding reproduces the conflict-free LDS pitches (72 / 40 elems).
// ---------------------------------------------------------------------------
__global__ __launch_bounds__(256) void attn_kernel(
    const unsigned short* __restrict__ qbf,    // [B,H,T,D]
    const unsigned short* __restrict__ kbf,    // [B,H,T,D]
    const unsigned short* __restrict__ vtb,    // [B,H,D,T]
    unsigned short* __restrict__ aout)         // bf16 [B,T,NX]
{
    __shared__ __align__(16) unsigned short sK[32 * 72];        // [key][d] pitch 72
    __shared__ __align__(16) unsigned short sVt[64 * 40];       // [d][key] pitch 40
    __shared__ __align__(16) unsigned short sP[8][16 * 40];     // per-wave P tile

    const int bh = blockIdx.x;
    const int q0 = blockIdx.y * 128;
    const int tid = threadIdx.x;
    const int w = tid >> 5, l = tid & 31;
    const int hh = l >> 4, ln = l & 15;
    const int qw = q0 + w * 16;

    const size_t base = (size_t)bh * T_ * HD_;
    const unsigned short* qp = qbf + base;
    const unsigned short* kp = kbf + base;
    const unsigned short* vp = vtb + base;     // [d][t], row stride T_

    Frag fq0, fq1;
    {
        const unsigned short* p = qp + (size_t)(qw + ln) * 64;
        fq0.q[0] = *(const u32x4*)(p + 8 * hh);
        fq0.q[1] = *(const u32x4*)(p + 16 + 8 * hh);
        fq1.q[0] = *(const u32x4*)(p + 32 + 8 * hh);
        fq1.q[1] = *(const u32x4*)(p + 48 + 8 * hh);
    }

    v8f O0 = {}, O1 = {}, O2 = {}, O3 = {};
    float mrow[8], lrow[8];
    #pragma unroll
    for (int r = 0; r < 8; ++r) { mrow[r] = 0.0f; lrow[r] = 0.0f; }

#ifdef HAVE_TDM
    const unsigned ldsK = (unsigned)(uintptr_t)(void*)sK;
    const unsigned ldsV = (unsigned)(uintptr_t)(void*)sVt;
#else
    const int skey = tid >> 3, sgrp = tid & 7;
#endif
    const int nIter = q0 / 32 + 4;

    for (int j = 0; j < nIter; ++j) {
        const int kbase = j * 32;
        __syncthreads();
#ifdef HAVE_TDM
        if (w == 0) {
            // K tile: 32 rows x 64 bf16, row stride 64 -> sK pitch 72
            tdm_load_2d(ldsK, kp + (size_t)kbase * 64, TDM_DW0_K, 64, 32, 64, 32, 64);
            // V tile: 64 rows (d) x 32 bf16 (t), row stride 2048 -> sVt pitch 40
            tdm_load_2d(ldsV, vp + kbase, TDM_DW0_V, 32, 64, 32, 64, 2048);
            __builtin_amdgcn_s_wait_tensorcnt(0);
        }
#else
        {
            u32x4 kv = *(const u32x4*)(kp + (size_t)(kbase + skey) * 64 + sgrp * 8);
            *(u32x4*)(sK + skey * 72 + sgrp * 8) = kv;
            // vp rows are [d][t]: thread copies 8 t-contiguous elems of one d row
            int d8 = tid >> 2, tg = tid & 3;  // 64 d rows x 4 grps of 8
            u32x4 vv = *(const u32x4*)(vp + (size_t)d8 * T_ + kbase + tg * 8);
            *(u32x4*)(sVt + d8 * 40 + tg * 8) = vv;
        }
#endif
        __syncthreads();

        float S0[8], S1[8];
        #pragma unroll
        for (int c = 0; c < 2; ++c) {
            Frag kb0, kb1;
            const unsigned short* p = sK + (c * 16 + ln) * 72 + 8 * hh;
            kb0.q[0] = *(const u32x4*)p;
            kb0.q[1] = *(const u32x4*)(p + 16);
            kb1.q[0] = *(const u32x4*)(p + 32);
            kb1.q[1] = *(const u32x4*)(p + 48);
            v8f acc = {};
            acc = __builtin_amdgcn_wmma_f32_16x16x32_bf16(false, fq0.v, false, kb0.v,
                                                          (short)0, acc, false, false);
            acc = __builtin_amdgcn_wmma_f32_16x16x32_bf16(false, fq1.v, false, kb1.v,
                                                          (short)0, acc, false, false);
            #pragma unroll
            for (int r = 0; r < 8; ++r) {
                int qg = qw + r + 8 * hh;
                int kg = kbase + c * 16 + ln;
                float s = acc[r] * 0.125f;               // 1/sqrt(64)
                s = (kg > qg) ? -10000.0f : s;           // reference mask value
                if (c == 0) S0[r] = s; else S1[r] = s;
            }
        }

        float alpha[8];
        #pragma unroll
        for (int r = 0; r < 8; ++r) {
            float mx = fmaxf(S0[r], S1[r]);
            mx = fmaxf(mx, __shfl_xor(mx, 1, 32));
            mx = fmaxf(mx, __shfl_xor(mx, 2, 32));
            mx = fmaxf(mx, __shfl_xor(mx, 4, 32));
            mx = fmaxf(mx, __shfl_xor(mx, 8, 32));
            float mnew = fmaxf(mrow[r], mx);
            alpha[r] = __expf(mrow[r] - mnew);
            mrow[r] = mnew;
            float p0 = __expf(S0[r] - mnew);
            float p1 = __expf(S1[r] - mnew);
            float rs = p0 + p1;
            rs += __shfl_xor(rs, 1, 32);
            rs += __shfl_xor(rs, 2, 32);
            rs += __shfl_xor(rs, 4, 32);
            rs += __shfl_xor(rs, 8, 32);
            lrow[r] = lrow[r] * alpha[r] + rs;
            O0[r] *= alpha[r]; O1[r] *= alpha[r];
            O2[r] *= alpha[r]; O3[r] *= alpha[r];
            int mr = r + 8 * hh;
            sP[w][mr * 40 + ln]      = f2bf(p0);
            sP[w][mr * 40 + 16 + ln] = f2bf(p1);
        }
        asm volatile("s_wait_dscnt 0x0" ::: "memory");   // intra-wave LDS RAW

        Frag fp;
        {
            const unsigned short* p = &sP[w][ln * 40];
            fp.q[0] = *(const u32x4*)(p + 8 * hh);
            fp.q[1] = *(const u32x4*)(p + 16 + 8 * hh);
        }
        #pragma unroll
        for (int dt = 0; dt < 4; ++dt) {
            Frag fv;
            const unsigned short* p = sVt + (dt * 16 + ln) * 40 + 8 * hh;
            fv.q[0] = *(const u32x4*)p;
            fv.q[1] = *(const u32x4*)(p + 16);
            v8f& O = (dt == 0) ? O0 : (dt == 1) ? O1 : (dt == 2) ? O2 : O3;
            O = __builtin_amdgcn_wmma_f32_16x16x32_bf16(false, fp.v, false, fv.v,
                                                        (short)0, O, false, false);
        }
    }

    const int bb = bh >> 4, head = bh & 15;
    #pragma unroll
    for (int r = 0; r < 8; ++r) {
        int tt = qw + r + 8 * hh;
        float inv = 1.0f / lrow[r];
        size_t rowoff = ((size_t)(bb * T_ + tt)) * NX_ + head * HD_;
        aout[rowoff + ln]      = f2bf(O0[r] * inv);
        aout[rowoff + 16 + ln] = f2bf(O1[r] * inv);
        aout[rowoff + 32 + ln] = f2bf(O2[r] * inv);
        aout[rowoff + 48 + ln] = f2bf(O3[r] * inv);
    }
}

// ---------------------------------------------------------------------------
// Output projection: [8192,1024] @ [1024,1024] + bias -> fp32 out
// ---------------------------------------------------------------------------
__global__ __launch_bounds__(256) void proj_gemm(
    const unsigned short* __restrict__ xa,     // [8192,1024] bf16
    const unsigned short* __restrict__ wt,     // [1024,1024] bf16 (transposed)
    const float* __restrict__ bias,            // [1024]
    float* __restrict__ out)                   // [8192,1024] fp32
{
    GEMM_PROLOGUE(xa, wt, 1024)

    #pragma unroll
    for (int sn = 0; sn < 2; ++sn) {
        const int n_g = n0 + wx * 32 + sn * 16 + ln;
        const float bi = bias[n_g];
        #pragma unroll
        for (int sm = 0; sm < 2; ++sm) {
            v8f acc = sm ? (sn ? acc11 : acc10) : (sn ? acc01 : acc00);
            #pragma unroll
            for (int r = 0; r < 8; ++r) {
                int m_g = m0 + wy * 32 + sm * 16 + r + 8 * hh;
                out[(size_t)m_g * 1024 + n_g] = acc[r] + bi;
            }
        }
    }
}

extern "C" void kernel_launch(void* const* d_in, const int* in_sizes, int n_in,
                              void* d_out, int out_size, void* d_ws, size_t ws_size,
                              hipStream_t stream) {
    (void)in_sizes; (void)n_in; (void)out_size; (void)ws_size;
    const float* x        = (const float*)d_in[0];
    // d_in[1] = causal mask, computed analytically instead
    const float* c_attn_w = (const float*)d_in[2];
    const float* c_attn_b = (const float*)d_in[3];
    const float* c_proj_w = (const float*)d_in[4];
    const float* c_proj_b = (const float*)d_in[5];

    float* out_a   = (float*)d_out;
    float* present = out_a + (size_t)MTOK * NX_;     // [2,B,H,T,D]

    char* ws = (char*)d_ws;
    size_t off = 0;
    auto alloc = [&](size_t bytes) -> void* {
        void* p = ws + off;
        off += (bytes + 255) & ~(size_t)255;
        return p;
    };
    unsigned short* x_bf     = (unsigned short*)alloc((size_t)MTOK * NX_ * 2);
    unsigned short* wqkvT_bf = (unsigned short*)alloc((size_t)3 * NX_ * NX_ * 2);
    unsigned short* wprojT_bf= (unsigned short*)alloc((size_t)NX_ * NX_ * 2);
    unsigned short* q_bf     = (unsigned short*)alloc((size_t)BHTD * 2);
    unsigned short* k_bf     = (unsigned short*)alloc((size_t)BHTD * 2);
    unsigned short* vt_bf    = (unsigned short*)alloc((size_t)BHTD * 2);
    unsigned short* a_bf     = (unsigned short*)alloc((size_t)MTOK * NX_ * 2);

    cvt_f32_to_bf16<<<2048, 256, 0, stream>>>(x, x_bf, MTOK * NX_);
    transpose_cvt<<<4096, 256, 0, stream>>>(c_attn_w, wqkvT_bf, 3 * NX_, 3 * NX_ * NX_);
    transpose_cvt<<<2048, 256, 0, stream>>>(c_proj_w, wprojT_bf, NX_, NX_ * NX_);

    dim3 g1(3 * NX_ / 128, MTOK / 64);  // (24, 128)
    qkv_gemm<<<g1, 256, 0, stream>>>(x_bf, wqkvT_bf, c_attn_b,
                                     q_bf, k_bf, vt_bf, present);

    dim3 g2(B_ * NH_, T_ / 128);        // (64, 16)
    attn_kernel<<<g2, 256, 0, stream>>>(q_bf, k_bf, vt_bf, a_bf);

    dim3 g3(NX_ / 128, MTOK / 64);      // (8, 128)
    proj_gemm<<<g3, 256, 0, stream>>>(a_bf, wprojT_bf, c_proj_b, out_a);
}
